// AttentionBlock_79628693667985
// MI455X (gfx1250) — matrix operand order repaired and checked by reading, not compiled, not run on hardware
//
#include <hip/hip_runtime.h>
#include <hip/hip_bf16.h>
#include <math.h>

// ---------------------------------------------------------------------------
// MI455X (gfx1250) attention block.
//   x:[2,2048,1024] f32.  All GEMMs on v_wmma_f32_16x16x32_bf16 (f32 accum),
//   tiles staged with global_load_async_to_lds_b128 (ASYNCcnt path),
//   streaming-softmax attention (no NxN score materialization), f32 LN.
// ---------------------------------------------------------------------------

typedef __attribute__((ext_vector_type(16))) __bf16          v16bf;
typedef __attribute__((ext_vector_type(16))) unsigned short  v16u;
typedef __attribute__((ext_vector_type(8)))  float           v8f;

#define SEQ   2048
#define DMODEL 1024
#define NHEAD 16
#define HDIM  64
#define MROWS 4096   // B*N

__device__ __forceinline__ unsigned short f2bf(float f) {
    unsigned int u = __builtin_bit_cast(unsigned int, f);
    u += 0x7FFFu + ((u >> 16) & 1u);       // round-to-nearest-even
    return (unsigned short)(u >> 16);
}

// ---- CDNA5 async global->LDS copy (CDNA5 ISA) ----------
// VDST = 32-bit LDS byte offset (flat addr low 32 bits), VADDR = 64-bit global.
__device__ __forceinline__ void async_load_b128(const unsigned short* lds_dst,
                                                const unsigned short* gsrc) {
    unsigned        l = (unsigned)(size_t)lds_dst;
    unsigned long long g = (unsigned long long)(size_t)gsrc;
    asm volatile("global_load_async_to_lds_b128 %0, %1, off"
                 :: "v"(l), "v"(g) : "memory");
}
__device__ __forceinline__ void wait_async0() {
    asm volatile("s_wait_asynccnt 0x0" ::: "memory");
}

// ---- WMMA fragment gathers (wave32, 16-bit; contiguous k order, same for A and B) ----------
// A 16x32 (MxK): lane L holds row M=L&15; half h=L>>4.
//   vgpr v: K0 = h*16 + v*2 ; elements K0,K0+1
__device__ __forceinline__ v16bf load_frag_a(const unsigned short* t, int ldk, int lane) {
    int m = lane & 15, h = lane >> 4;
    v16u r;
#pragma unroll
    for (int v = 0; v < 8; ++v) {
        int k0 = h * 16 + v * 2;
        r[2 * v]     = t[m * ldk + k0];
        r[2 * v + 1] = t[m * ldk + k0 + 1];
    }
    return __builtin_bit_cast(v16bf, r);
}

// B 32x16 (KxN), tile stored row-per-N: t[n*ldk + k].
__device__ __forceinline__ v16bf load_frag_b(const unsigned short* t, int ldk, int lane) {
    int n = lane & 15, h = lane >> 4;
    v16u r;
#pragma unroll
    for (int v = 0; v < 8; ++v) {
        int k0 = h * 16 + v * 2;
        r[2 * v]     = t[n * ldk + k0];
        r[2 * v + 1] = t[n * ldk + k0 + 1];
    }
    return __builtin_bit_cast(v16bf, r);
}

// B 32x16 from a K-major tile: t[k*ldn + n]
__device__ __forceinline__ v16bf load_frag_bt(const unsigned short* t, int ldn, int lane) {
    int n = lane & 15, h = lane >> 4;
    v16u r;
#pragma unroll
    for (int v = 0; v < 8; ++v) {
        int k0 = h * 16 + v * 2;
        r[2 * v]     = t[(k0)     * ldn + n];
        r[2 * v + 1] = t[(k0 + 1) * ldn + n];
    }
    return __builtin_bit_cast(v16bf, r);
}

__device__ __forceinline__ v8f wmma_bf16(v16bf a, v16bf b, v8f c) {
    return __builtin_amdgcn_wmma_f32_16x16x32_bf16(false, a, false, b, (short)0, c, false, false);
}

// 16-lane butterfly reductions (rows live in half-waves for C/D layout)
__device__ __forceinline__ float rmax16(float v) {
#pragma unroll
    for (int o = 1; o < 16; o <<= 1) v = fmaxf(v, __shfl_xor(v, o, 32));
    return v;
}
__device__ __forceinline__ float rsum16(float v) {
#pragma unroll
    for (int o = 1; o < 16; o <<= 1) v += __shfl_xor(v, o, 32);
    return v;
}

// ---------------------------------------------------------------------------
// Kernel 0: f32 -> bf16 convert
// ---------------------------------------------------------------------------
__global__ void cvt_bf16_kernel(const float* __restrict__ src,
                                unsigned short* __restrict__ dst, int n) {
    int i = blockIdx.x * blockDim.x + threadIdx.x;
    if (i < n) dst[i] = f2bf(src[i]);
}

// ---------------------------------------------------------------------------
// Kernel 1/3: C[m][n] = sum_k A[m][k]*W[n][k] + bias[n]
//  A: MxK bf16 row-major, W: NxK bf16 row-major (torch Linear weight layout)
//  Block: 128 thr (4 waves) -> 64x64 tile, each wave 16x64. K-step 32.
//  Tiles staged via async global->LDS B128 (80-byte padded rows, 16B aligned).
// ---------------------------------------------------------------------------
template <bool OUT_BF16>
__global__ __launch_bounds__(128)
void gemm_bf16_kernel(const unsigned short* __restrict__ A,
                      const unsigned short* __restrict__ W,
                      const float* __restrict__ bias,
                      void* __restrict__ Cout, int M, int N, int K) {
    const int LDT = 40;  // u16 stride: 80 B rows -> 16B-aligned b128 chunks
    __shared__ __align__(16) unsigned short At[64 * 40];
    __shared__ __align__(16) unsigned short Wt[64 * 40];

    const int tid = threadIdx.x, lane = tid & 31, w = tid >> 5;
    const int m0 = blockIdx.y * 64, n0 = blockIdx.x * 64;

    v8f acc[4] = {};
    for (int k0 = 0; k0 < K; k0 += 32) {
        __syncthreads();
        // 64x32 u16 per tile = 256 x 16B chunks; 2 per thread per tile
#pragma unroll
        for (int i = 0; i < 2; ++i) {
            int lin = i * 128 + tid;
            int row = lin >> 2, c = lin & 3;          // 4 chunks per 32-u16 row
            async_load_b128(&At[row * LDT + c * 8],
                            A + (size_t)(m0 + row) * K + k0 + c * 8);
            async_load_b128(&Wt[row * LDT + c * 8],
                            W + (size_t)(n0 + row) * K + k0 + c * 8);
        }
        wait_async0();
        __syncthreads();
        v16bf a = load_frag_a(At + (w * 16) * LDT, LDT, lane);
#pragma unroll
        for (int nt = 0; nt < 4; ++nt) {
            v16bf b = load_frag_b(Wt + (nt * 16) * LDT, LDT, lane);
            acc[nt] = wmma_bf16(a, b, acc[nt]);
        }
    }
    const int hh = lane >> 4, nn = lane & 15;
#pragma unroll
    for (int nt = 0; nt < 4; ++nt) {
        int col = n0 + nt * 16 + nn;
        float bv = bias[col];
#pragma unroll
        for (int r = 0; r < 8; ++r) {
            int row = m0 + w * 16 + r + 8 * hh;
            float val = acc[nt][r] + bv;
            if constexpr (OUT_BF16)
                ((unsigned short*)Cout)[(size_t)row * N + col] = f2bf(val);
            else
                ((float*)Cout)[(size_t)row * N + col] = val;
        }
    }
}

// ---------------------------------------------------------------------------
// Kernel 2: flash attention.  grid=(B*H, N/64), block=128 (4 waves x 16 rows)
//  K/V staged 32x64 bf16 in LDS (async B128) shared by 4 waves; f32 softmax.
// ---------------------------------------------------------------------------
__global__ __launch_bounds__(128)
void flash_attn_kernel(const unsigned short* __restrict__ Qb,
                       const unsigned short* __restrict__ Kb,
                       const unsigned short* __restrict__ Vb,
                       const float* __restrict__ ascale,
                       unsigned short* __restrict__ Ob) {
    __shared__ __align__(16) unsigned short Kt[32 * 64];
    __shared__ __align__(16) unsigned short Vt[32 * 64];
    __shared__ __align__(16) unsigned short Pt[4 * 16 * 32];

    const int tid = threadIdx.x, lane = tid & 31, w = tid >> 5;
    const int b = blockIdx.x / NHEAD, h = blockIdx.x % NHEAD;
    const int rowbase = b * SEQ + blockIdx.y * 64 + w * 16;
    const int bbase   = b * SEQ;
    const int hh = lane >> 4, nn = lane & 15;
    const float sc = 0.125f * ascale[h];          // HD^-0.5 / TEMP

    // Q fragments (16 rows x 64 hd), held in registers all loop long
    const unsigned short* qptr = Qb + (size_t)rowbase * DMODEL + h * HDIM;
    v16bf qa0 = load_frag_a(qptr,      DMODEL, lane);
    v16bf qa1 = load_frag_a(qptr + 32, DMODEL, lane);

    float m_run[8], l_run[8];
#pragma unroll
    for (int r = 0; r < 8; ++r) { m_run[r] = -1e30f; l_run[r] = 0.f; }
    v8f o[4] = {};

    for (int j0 = 0; j0 < SEQ; j0 += 32) {
        __syncthreads();
        // 32x64 u16 per tile = 256 x 16B chunks; 2 per thread per tile
#pragma unroll
        for (int i = 0; i < 2; ++i) {
            int lin = i * 128 + tid;
            int row = lin >> 3, c = lin & 7;          // 8 chunks per 64-u16 row
            size_t g = (size_t)(bbase + j0 + row) * DMODEL + h * HDIM + c * 8;
            async_load_b128(&Kt[row * 64 + c * 8], Kb + g);
            async_load_b128(&Vt[row * 64 + c * 8], Vb + g);
        }
        wait_async0();
        __syncthreads();

        // scores: two 16x16 tiles covering 32 keys
        v8f s0 = {}, s1 = {};
        s0 = wmma_bf16(qa0, load_frag_b(Kt,                64, lane), s0);
        s0 = wmma_bf16(qa1, load_frag_b(Kt + 32,           64, lane), s0);
        s1 = wmma_bf16(qa0, load_frag_b(Kt + 16 * 64,      64, lane), s1);
        s1 = wmma_bf16(qa1, load_frag_b(Kt + 16 * 64 + 32, 64, lane), s1);

        float p0[8], p1[8], alpha[8];
#pragma unroll
        for (int r = 0; r < 8; ++r) {
            float v0 = s0[r] * sc, v1 = s1[r] * sc;
            float mx = rmax16(fmaxf(v0, v1));
            float mn = fmaxf(m_run[r], mx);
            p0[r] = __expf(v0 - mn);
            p1[r] = __expf(v1 - mn);
            float rs = rsum16(p0[r] + p1[r]);
            alpha[r] = __expf(m_run[r] - mn);
            l_run[r] = l_run[r] * alpha[r] + rs;
            m_run[r] = mn;
        }
#pragma unroll
        for (int r = 0; r < 8; ++r) {
            o[0][r] *= alpha[r]; o[1][r] *= alpha[r];
            o[2][r] *= alpha[r]; o[3][r] *= alpha[r];
        }

        // re-layout P (C-layout f32) -> A-layout bf16 via per-wave LDS slab
        unsigned short* Pw = Pt + w * 512;
#pragma unroll
        for (int r = 0; r < 8; ++r) {
            int row = r + 8 * hh;
            Pw[row * 32 + nn]      = f2bf(p0[r]);
            Pw[row * 32 + 16 + nn] = f2bf(p1[r]);
        }
        v16bf ap = load_frag_a(Pw, 32, lane);
        o[0] = wmma_bf16(ap, load_frag_bt(Vt,      64, lane), o[0]);
        o[1] = wmma_bf16(ap, load_frag_bt(Vt + 16, 64, lane), o[1]);
        o[2] = wmma_bf16(ap, load_frag_bt(Vt + 32, 64, lane), o[2]);
        o[3] = wmma_bf16(ap, load_frag_bt(Vt + 48, 64, lane), o[3]);
    }

#pragma unroll
    for (int nt = 0; nt < 4; ++nt)
#pragma unroll
        for (int r = 0; r < 8; ++r) {
            int row = rowbase + r + 8 * hh;
            int col = h * HDIM + nt * 16 + nn;
            Ob[(size_t)row * DMODEL + col] = f2bf(o[nt][r] / l_run[r]);
        }
}

// ---------------------------------------------------------------------------
// Kernel 4: out = LayerNorm(x + proj) * gamma + beta   (one block per row)
// ---------------------------------------------------------------------------
__global__ __launch_bounds__(256)
void resid_ln_kernel(const float* __restrict__ x, const float* __restrict__ proj,
                     const float* __restrict__ gamma, const float* __restrict__ beta,
                     float* __restrict__ out) {
    __shared__ float sbuf[256], ssbuf[256];
    const int row = blockIdx.x;
    const float* xr = x    + (size_t)row * DMODEL;
    const float* pr = proj + (size_t)row * DMODEL;
    float*       orow = out + (size_t)row * DMODEL;

    float y[4], s = 0.f, ss = 0.f;
#pragma unroll
    for (int i = 0; i < 4; ++i) {
        int c = threadIdx.x + i * 256;
        float v = xr[c] + pr[c];
        y[i] = v; s += v; ss += v * v;
    }
    sbuf[threadIdx.x] = s; ssbuf[threadIdx.x] = ss;
    __syncthreads();
    for (int st = 128; st > 0; st >>= 1) {
        if (threadIdx.x < st) {
            sbuf[threadIdx.x]  += sbuf[threadIdx.x + st];
            ssbuf[threadIdx.x] += ssbuf[threadIdx.x + st];
        }
        __syncthreads();
    }
    const float mu  = sbuf[0] * (1.0f / DMODEL);
    const float var = ssbuf[0] * (1.0f / DMODEL) - mu * mu;
    const float inv = rsqrtf(var + 1e-5f);
#pragma unroll
    for (int i = 0; i < 4; ++i) {
        int c = threadIdx.x + i * 256;
        orow[c] = (y[i] - mu) * inv * gamma[c] + beta[c];
    }
}

// ---------------------------------------------------------------------------
extern "C" void kernel_launch(void* const* d_in, const int* in_sizes, int n_in,
                              void* d_out, int out_size, void* d_ws, size_t ws_size,
                              hipStream_t stream) {
    const float* x      = (const float*)d_in[0];
    const float* ascale = (const float*)d_in[1];
    const float* Wq = (const float*)d_in[2];  const float* bq = (const float*)d_in[3];
    const float* Wk = (const float*)d_in[4];  const float* bk = (const float*)d_in[5];
    const float* Wv = (const float*)d_in[6];  const float* bv = (const float*)d_in[7];
    const float* Wo = (const float*)d_in[8];  const float* bo = (const float*)d_in[9];
    const float* gamma = (const float*)d_in[10];
    const float* beta  = (const float*)d_in[11];

    // workspace layout (u16 elements unless noted); total ~64 MiB
    unsigned short* xb   = (unsigned short*)d_ws;              // 4096x1024
    unsigned short* wqb  = xb  + (size_t)MROWS * DMODEL;       // 1024x1024
    unsigned short* wkb  = wqb + (size_t)DMODEL * DMODEL;
    unsigned short* wvb  = wkb + (size_t)DMODEL * DMODEL;
    unsigned short* wob  = wvb + (size_t)DMODEL * DMODEL;
    unsigned short* Qb   = wob + (size_t)DMODEL * DMODEL;      // 4096x1024
    unsigned short* Kb   = Qb  + (size_t)MROWS * DMODEL;
    unsigned short* Vb   = Kb  + (size_t)MROWS * DMODEL;
    unsigned short* attb = Vb  + (size_t)MROWS * DMODEL;
    float*          projf = (float*)(attb + (size_t)MROWS * DMODEL); // 4096x1024 f32

    const int nX = MROWS * DMODEL, nW = DMODEL * DMODEL;
    cvt_bf16_kernel<<<nX / 256, 256, 0, stream>>>(x,  xb,  nX);
    cvt_bf16_kernel<<<nW / 256, 256, 0, stream>>>(Wq, wqb, nW);
    cvt_bf16_kernel<<<nW / 256, 256, 0, stream>>>(Wk, wkb, nW);
    cvt_bf16_kernel<<<nW / 256, 256, 0, stream>>>(Wv, wvb, nW);
    cvt_bf16_kernel<<<nW / 256, 256, 0, stream>>>(Wo, wob, nW);

    dim3 ggrid(DMODEL / 64, MROWS / 64);   // (N/64, M/64)
    gemm_bf16_kernel<true ><<<ggrid, 128, 0, stream>>>(xb, wqb, bq, Qb, MROWS, DMODEL, DMODEL);
    gemm_bf16_kernel<true ><<<ggrid, 128, 0, stream>>>(xb, wkb, bk, Kb, MROWS, DMODEL, DMODEL);
    gemm_bf16_kernel<true ><<<ggrid, 128, 0, stream>>>(xb, wvb, bv, Vb, MROWS, DMODEL, DMODEL);

    flash_attn_kernel<<<dim3(2 * NHEAD, SEQ / 64), 128, 0, stream>>>(Qb, Kb, Vb, ascale, attb);

    gemm_bf16_kernel<false><<<ggrid, 128, 0, stream>>>(attb, wob, bo, projf, MROWS, DMODEL, DMODEL);

    resid_ln_kernel<<<MROWS, 256, 0, stream>>>(x, projf, gamma, beta, (float*)d_out);
}
